// LocalPMFSBlock_AP_38165079392804
// MI455X (gfx1250) — matrix operand order repaired and checked by reading, not compiled, not run on hardware
//
#include <hip/hip_runtime.h>

typedef __attribute__((ext_vector_type(16))) _Float16 v16h;
typedef __attribute__((ext_vector_type(8)))  _Float16 v8h;
typedef __attribute__((ext_vector_type(8)))  float    v8f;
typedef __attribute__((ext_vector_type(4)))  int      v4i;

#define NVOX  110592      // 48*48*48
#define NT    64          // voxels per tile for 1x1-conv kernels
#define XSTR  104         // LDS row stride (halves) for [voxel][channel] tiles
#define NTILE 1728        // NVOX / NT
#define EPSBN 1e-5f

#if defined(__gfx1250__) && __has_builtin(__builtin_amdgcn_global_load_async_to_lds_b128)
#define HAVE_ASYNC_LDS 1
#else
#define HAVE_ASYNC_LDS 0
#endif

struct Ptrs {
  const float* x;
  const float* w[7];      // ch_Wq, ch_Wk, ch_Wv, sp_Wq, sp_Wk, sp_Wv, sp_out (conv weights)
  const float* gg[7];
  const float* bb[7];
  const float* mm[7];
  const float* vv[7];
  const float* score_W;
  const float* score_b;
  const float* ln_g;
  const float* ln_b;
  _Float16* ch16;         // ch_out f16, voxel-major: [2][NVOX][96]
  _Float16* sp16;         // sp_out f16, voxel-major: [2][NVOX][96]
  _Float16* wqP;          // WMMA-A-fragment packed 96x96 weights
  _Float16* wvP;
  _Float16* sqP;
  _Float16* skP;
  _Float16* svP;
  _Float16* w3P;          // 27 taps x packed 96x96
  float* sb;              // [7][192]: s[96], t[96] per conv block
  float* red;             // [0..1] sum_e, [2..193] sum_qe, [194..257] spk acc,
                          // [258..449] ch gate, [450..513] sp softmax w
  float* out;
};

// ---- helpers -------------------------------------------------------------

__device__ __forceinline__ v16h ld_b16(const _Float16* p) {
  v8h lo = *(const v8h*)p;
  v8h hi = *(const v8h*)(p + 8);
  v16h r;
#pragma unroll
  for (int j = 0; j < 8; ++j) { r[j] = lo[j]; r[j + 8] = hi[j]; }
  return r;
}

// 16-byte global->LDS copy: async DMA when available, sync fallback otherwise.
__device__ __forceinline__ void copy8_g2l(const _Float16* gp, _Float16* lp) {
#if HAVE_ASYNC_LDS
  __builtin_amdgcn_global_load_async_to_lds_b128(
      (__attribute__((address_space(1))) v4i*)(void*)gp,
      (__attribute__((address_space(3))) v4i*)(void*)lp, 0, 0);
#else
  *(v8h*)lp = *(const v8h*)gp;
#endif
}

__device__ __forceinline__ void async_lds_fence() {
#if HAVE_ASYNC_LDS
  asm volatile("s_wait_asynccnt 0x0" ::: "memory");
#endif
}

// One 16x16 output tile of a 96x96 (A, frag-packed) @ 96xNT (B, LDS voxel-major) GEMM.
__device__ __forceinline__ v8f gemm96(const _Float16* __restrict__ wp,
                                      const _Float16* Xl,
                                      int mt, int col, int hioff, int lane) {
  v8f acc = {0.f, 0.f, 0.f, 0.f, 0.f, 0.f, 0.f, 0.f};
#pragma unroll
  for (int kt = 0; kt < 3; ++kt) {
    v16h a = ld_b16(wp + ((mt * 3 + kt) * 32 + lane) * 16);
    v16h b = ld_b16(Xl + col * XSTR + kt * 32 + hioff);
    acc = __builtin_amdgcn_wmma_f32_16x16x32_f16(false, a, false, b,
                                                 (short)0, acc, false, false);
  }
  return acc;
}

// Pack a row-major 96x96 fp32 weight into WMMA A-fragment order (f16).
__device__ __forceinline__ void pack96(const float* W, _Float16* dst, int tid, int nthr) {
  for (int idx = tid; idx < 9216; idx += nthr) {
    int j = idx & 15, lane = (idx >> 4) & 31, f = idx >> 9;  // f in [0,18)
    int mt = f / 3, kt = f % 3;
    int mrow = mt * 16 + (lane & 15);
    int kbase = j;
    int k = kt * 32 + kbase + ((lane & 16) ? 16 : 0);
    dst[idx] = (_Float16)W[mrow * 96 + k];
  }
}

// ---- K0: weight prep + BN fold + accumulator zeroing ---------------------

__global__ void k0_prep(Ptrs P) {
  int tid = threadIdx.x;
  pack96(P.w[0], P.wqP, tid, 256);
  pack96(P.w[2], P.wvP, tid, 256);
  pack96(P.w[3], P.sqP, tid, 256);
  pack96(P.w[4], P.skP, tid, 256);
  pack96(P.w[5], P.svP, tid, 256);
  // 3x3x3 conv: [o][i][z][y][x] -> per-tap A-fragment packs
  for (int idx = tid; idx < 248832; idx += 256) {
    int j = idx & 15, lane = (idx >> 4) & 31, f = idx >> 9;  // f in [0,486)
    int kt = f % 3, mt = (f / 3) % 6, tap = f / 18;
    int mrow = mt * 16 + (lane & 15);
    int kbase = j;
    int k = kt * 32 + kbase + ((lane & 16) ? 16 : 0);
    P.w3P[idx] = (_Float16)P.w[6][(mrow * 96 + k) * 27 + tap];
  }
  // BN scale/bias: y = conv*s + t
  for (int i = tid; i < 7 * 96; i += 256) {
    int blk = i / 96, o = i % 96;
    int cnt = (blk == 1) ? 1 : 96;
    if (o < cnt) {
      float s = P.gg[blk][o] * rsqrtf(P.vv[blk][o] + EPSBN);
      P.sb[blk * 192 + o] = s;
      P.sb[blk * 192 + 96 + o] = P.bb[blk][o] - P.mm[blk][o] * s;
    }
  }
  for (int i = tid; i < 258; i += 256) P.red[i] = 0.f;
}

// ---- K1: ch_Q GEMM + ch_K logits + softmax-weighted reduction ------------

__global__ void k1_chq(Ptrs P) {
  __shared__ _Float16 Xl[NT * XSTR];
  __shared__ float e_l[NT];
  __shared__ float qacc[96];
  int tid = threadIdx.x;
  int b = blockIdx.x / NTILE;
  int n0 = (blockIdx.x % NTILE) * NT;
  if (tid < 96) qacc[tid] = 0.f;
  for (int idx = tid; idx < 96 * NT; idx += 256) {
    int c = idx >> 6, i = idx & 63;
    Xl[i * XSTR + c] = (_Float16)P.x[(b * 96 + c) * NVOX + n0 + i];
  }
  __syncthreads();
  if (tid < NT) {
    const float* wk = P.w[1];
    float a = 0.f;
    for (int c = 0; c < 96; ++c) a += (float)Xl[tid * XSTR + c] * wk[c];
    float kv = fmaxf(a * P.sb[1 * 192 + 0] + P.sb[1 * 192 + 96 + 0], 0.f);
    e_l[tid] = __expf(kv);
  }
  __syncthreads();
  int wave = tid >> 5, lane = tid & 31, lrow = lane & 15;
  int hioff = (lane & 16) ? 16 : 0, hi8 = (lane & 16) ? 8 : 0;
  const float* sq = P.sb + 0 * 192;
  const float* tq = sq + 96;
  for (int s = wave; s < 24; s += 8) {
    int mt = s % 6, nc = s / 6;
    int col = nc * 16 + lrow;
    v8f acc = gemm96(P.wqP, Xl, mt, col, hioff, lane);
    float e = e_l[col];
#pragma unroll
    for (int r = 0; r < 8; ++r) {
      int m = mt * 16 + hi8 + r;
      float y = fmaxf(acc[r] * sq[m] + tq[m], 0.f);
      atomicAdd(&qacc[m], y * e);
    }
  }
  __syncthreads();
  if (tid == 0) {
    float se = 0.f;
    for (int i = 0; i < NT; ++i) se += e_l[i];
    atomicAdd(&P.red[b], se);
  }
  if (tid < 96) atomicAdd(&P.red[2 + b * 96 + tid], qacc[tid]);
}

// ---- K2: channel gate (score matvec + LayerNorm + sigmoid) ---------------

__global__ void k2_gate(Ptrs P) {
  __shared__ float Z[96];
  __shared__ float Zs[96];
  __shared__ float rl[128];
  int tid = threadIdx.x;
  int b = blockIdx.x;
  if (tid < 96) Z[tid] = P.red[2 + b * 96 + tid] / P.red[b];
  __syncthreads();
  if (tid < 96) {
    float a = P.score_b[tid];
    for (int k = 0; k < 96; ++k) a += P.score_W[tid * 96 + k] * Z[k];
    Zs[tid] = a;
  }
  __syncthreads();
  rl[tid] = (tid < 96) ? Zs[tid] : 0.f;
  __syncthreads();
  for (int st = 64; st > 0; st >>= 1) {
    if (tid < st) rl[tid] += rl[tid + st];
    __syncthreads();
  }
  float mu = rl[0] / 96.f;
  __syncthreads();
  rl[tid] = (tid < 96) ? (Zs[tid] - mu) * (Zs[tid] - mu) : 0.f;
  __syncthreads();
  for (int st = 64; st > 0; st >>= 1) {
    if (tid < st) rl[tid] += rl[tid + st];
    __syncthreads();
  }
  float var = rl[0] / 96.f;
  if (tid < 96) {
    float zn = (Zs[tid] - mu) * rsqrtf(var + EPSBN) * P.ln_g[tid] + P.ln_b[tid];
    P.red[258 + b * 96 + tid] = 1.f / (1.f + __expf(-zn));
  }
}

// ---- K3: ch_V GEMM + gate -> ch16[n][c]; fused sp_K GEMM + reduction -----

__global__ void k3_chv(Ptrs P) {
  __shared__ _Float16 Xl[NT * XSTR];
  __shared__ _Float16 Yl[NT * XSTR];
  __shared__ float kacc[32];
  int tid = threadIdx.x;
  int b = blockIdx.x / NTILE;
  int n0 = (blockIdx.x % NTILE) * NT;
  if (tid < 32) kacc[tid] = 0.f;
  for (int idx = tid; idx < 96 * NT; idx += 256) {
    int c = idx >> 6, i = idx & 63;
    Xl[i * XSTR + c] = (_Float16)P.x[(b * 96 + c) * NVOX + n0 + i];
  }
  __syncthreads();
  int wave = tid >> 5, lane = tid & 31, lrow = lane & 15;
  int hioff = (lane & 16) ? 16 : 0, hi8 = (lane & 16) ? 8 : 0;
  const float* sv = P.sb + 2 * 192;
  const float* tv = sv + 96;
  const float* gate = P.red + 258 + b * 96;
  for (int s = wave; s < 24; s += 8) {
    int mt = s % 6, nc = s / 6, col = nc * 16 + lrow;
    v8f acc = gemm96(P.wvP, Xl, mt, col, hioff, lane);
#pragma unroll
    for (int r = 0; r < 8; ++r) {
      int m = mt * 16 + hi8 + r;
      float y = fmaxf(acc[r] * sv[m] + tv[m], 0.f) * gate[m];
      Yl[col * XSTR + m] = (_Float16)y;
    }
  }
  __syncthreads();
  for (int idx = tid; idx < 96 * NT; idx += 256) {
    int c = idx % 96, i = idx / 96;
    P.ch16[((size_t)b * NVOX + n0 + i) * 96 + c] = Yl[i * XSTR + c];
  }
  const float* sk = P.sb + 4 * 192;
  const float* tk = sk + 96;
  for (int s = wave; s < 24; s += 8) {
    int mt = s % 6, nc = s / 6, col = nc * 16 + lrow;
    v8f acc = gemm96(P.skP, Yl, mt, col, hioff, lane);
#pragma unroll
    for (int r = 0; r < 8; ++r) {
      int m = mt * 16 + hi8 + r;
      float y = fmaxf(acc[r] * sk[m] + tk[m], 0.f);
      atomicAdd(&kacc[m & 31], y);
    }
  }
  __syncthreads();
  if (tid < 32) atomicAdd(&P.red[194 + b * 32 + tid], kacc[tid]);
}

// ---- K5: softmax over 32 channel means -----------------------------------

__global__ void k5_smax(Ptrs P) {
  __shared__ float vl[32];
  __shared__ float mx, sm;
  int tid = threadIdx.x;
  int b = blockIdx.x;
  float m = P.red[194 + b * 32 + tid] / (3.f * (float)NVOX);
  vl[tid] = m;
  __syncthreads();
  if (tid == 0) {
    float a = vl[0];
    for (int i = 1; i < 32; ++i) a = fmaxf(a, vl[i]);
    mx = a;
  }
  __syncthreads();
  float e = __expf(m - mx);
  vl[tid] = e;
  __syncthreads();
  if (tid == 0) {
    float a = 0.f;
    for (int i = 0; i < 32; ++i) a += vl[i];
    sm = a;
  }
  __syncthreads();
  P.red[450 + b * 32 + tid] = e / sm;
}

// ---- K6: sp_Q / sp_V GEMMs + spatial gating -> sp16[n][c] ----------------

__global__ void k6_sp(Ptrs P) {
  __shared__ _Float16 Xl[NT * XSTR];
  __shared__ _Float16 Ql[NT * XSTR];
  __shared__ _Float16 Vl[NT * XSTR];
  __shared__ float zl[3 * NT];
  __shared__ float wl[32];
  int tid = threadIdx.x;
  int b = blockIdx.x / NTILE;
  int n0 = (blockIdx.x % NTILE) * NT;
  if (tid < 32) wl[tid] = P.red[450 + b * 32 + tid];
  // async tile fill: ch16 is voxel-major so each 16B chunk is contiguous
  for (int idx = tid; idx < NT * 12; idx += 256) {
    int ck = idx % 12, i = idx / 12;
    copy8_g2l(P.ch16 + ((size_t)b * NVOX + n0 + i) * 96 + ck * 8,
              &Xl[i * XSTR + ck * 8]);
  }
  async_lds_fence();
  __syncthreads();
  int wave = tid >> 5, lane = tid & 31, lrow = lane & 15;
  int hioff = (lane & 16) ? 16 : 0, hi8 = (lane & 16) ? 8 : 0;
  const float* sQ = P.sb + 3 * 192;
  const float* tQ = sQ + 96;
  const float* sV = P.sb + 5 * 192;
  const float* tV = sV + 96;
  for (int s = wave; s < 24; s += 8) {
    int mt = s % 6, nc = s / 6, col = nc * 16 + lrow;
    v8f aq = gemm96(P.sqP, Xl, mt, col, hioff, lane);
    v8f av = gemm96(P.svP, Xl, mt, col, hioff, lane);
#pragma unroll
    for (int r = 0; r < 8; ++r) {
      int m = mt * 16 + hi8 + r;
      Ql[col * XSTR + m] = (_Float16)fmaxf(aq[r] * sQ[m] + tQ[m], 0.f);
      Vl[col * XSTR + m] = (_Float16)fmaxf(av[r] * sV[m] + tV[m], 0.f);
    }
  }
  __syncthreads();
  if (tid < 192) {
    int br = tid / 64, i = tid % 64;
    float z = 0.f;
    for (int ck = 0; ck < 32; ++ck) z += wl[ck] * (float)Ql[i * XSTR + br * 32 + ck];
    zl[br * 64 + i] = 1.f / (1.f + __expf(-z));
  }
  __syncthreads();
  for (int idx = tid; idx < 96 * NT; idx += 256) {
    int c = idx % 96, i = idx / 96;
    float o = (float)Vl[i * XSTR + c] * zl[(c >> 5) * 64 + i];
    P.sp16[((size_t)b * NVOX + n0 + i) * 96 + c] = (_Float16)o;
  }
}

// ---- K7: 3x3x3 conv as implicit WMMA GEMM + BN + ReLU --------------------
// Block covers 2 output rows (h0,h0+1) x 48 voxels; 6 waves, each owning
// 2 M-tiles x 3 N-tiles (one B fragment feeds two WMMAs).

__global__ void k7_conv3(Ptrs P) {
  __shared__ _Float16 Xh[12 * 50 * 40];  // [dz*4+hh][x halo][32ch padded to 40]
  int tid = threadIdx.x;
  int b = blockIdx.x / 1152;
  int rem = blockIdx.x % 1152;
  int d = rem / 24, h0 = (rem % 24) * 2;
  int wave = tid >> 5, lane = tid & 31, lrow = lane & 15;
  int hioff = (lane & 16) ? 16 : 0, hi8 = (lane & 16) ? 8 : 0;
  int rowSel = wave / 3;          // which of the 2 output rows
  int mt0 = (wave % 3) * 2;       // first of the wave's two M-tiles
  v8f acc[2][3];
#pragma unroll
  for (int u = 0; u < 2; ++u)
#pragma unroll
    for (int nc = 0; nc < 3; ++nc)
      acc[u][nc] = (v8f){0.f, 0.f, 0.f, 0.f, 0.f, 0.f, 0.f, 0.f};
  for (int kt = 0; kt < 3; ++kt) {
    __syncthreads();
    // halo tile fill: 12 planes x 50 x-positions x 4 16B chunks of 32 channels
    for (int idx = tid; idx < 12 * 50 * 4; idx += 192) {
      int ck = idx & 3;
      int t = idx >> 2;
      int q = t / 50, xin = t - q * 50;
      int din = d + (q >> 2) - 1, hin = h0 + (q & 3) - 1, xg = xin - 1;
      _Float16* lp = &Xh[(q * 50 + xin) * 40 + ck * 8];
      if ((unsigned)din < 48u && (unsigned)hin < 48u && (unsigned)xg < 48u) {
        const _Float16* gp = P.sp16 +
            ((size_t)b * NVOX + (din * 48 + hin) * 48 + xg) * 96 + kt * 32 + ck * 8;
        copy8_g2l(gp, lp);
      } else {
        *(v8h*)lp = (v8h)(_Float16)0.f;
      }
    }
    async_lds_fence();
    __syncthreads();
    for (int tap = 0; tap < 27; ++tap) {
      int dz = tap / 9, dy = (tap / 3) % 3, dx = tap % 3;
      int q = dz * 4 + dy + rowSel;
      if (tap < 26)
        __builtin_prefetch(P.w3P + (size_t)((tap + 1) * 18 + mt0 * 3 + kt) * 512 + lane * 16, 0, 1);
      v16h a0 = ld_b16(P.w3P + (size_t)(tap * 18 + mt0 * 3 + kt) * 512 + lane * 16);
      v16h a1 = ld_b16(P.w3P + (size_t)(tap * 18 + (mt0 + 1) * 3 + kt) * 512 + lane * 16);
#pragma unroll
      for (int nc = 0; nc < 3; ++nc) {
        v16h bm = ld_b16(&Xh[(q * 50 + nc * 16 + lrow + dx) * 40 + hioff]);
        acc[0][nc] = __builtin_amdgcn_wmma_f32_16x16x32_f16(false, a0, false, bm,
                                                            (short)0, acc[0][nc], false, false);
        acc[1][nc] = __builtin_amdgcn_wmma_f32_16x16x32_f16(false, a1, false, bm,
                                                            (short)0, acc[1][nc], false, false);
      }
    }
  }
  const float* so = P.sb + 6 * 192;
  const float* to = so + 96;
  int h = h0 + rowSel;
  size_t nbase = (size_t)b * 96 * NVOX + (size_t)(d * 48 + h) * 48;
#pragma unroll
  for (int u = 0; u < 2; ++u)
#pragma unroll
    for (int nc = 0; nc < 3; ++nc)
#pragma unroll
      for (int r = 0; r < 8; ++r) {
        int m = (mt0 + u) * 16 + hi8 + r;
        P.out[nbase + (size_t)m * NVOX + nc * 16 + lrow] =
            fmaxf(acc[u][nc][r] * so[m] + to[m], 0.f);
      }
}

// ---- host ----------------------------------------------------------------

extern "C" void kernel_launch(void* const* d_in, const int* in_sizes, int n_in,
                              void* d_out, int out_size, void* d_ws, size_t ws_size,
                              hipStream_t stream) {
  (void)in_sizes; (void)n_in; (void)out_size; (void)ws_size;
  char* w = (char*)d_ws;
  auto alloc = [&](size_t bytes) {
    char* p = w;
    w += (bytes + 255) & ~(size_t)255;
    return p;
  };
  Ptrs P;
  P.x = (const float*)d_in[0];
  const int base[7] = {1, 6, 11, 20, 25, 30, 35};
  for (int i = 0; i < 7; ++i) {
    P.w[i]  = (const float*)d_in[base[i]];
    P.gg[i] = (const float*)d_in[base[i] + 1];
    P.bb[i] = (const float*)d_in[base[i] + 2];
    P.mm[i] = (const float*)d_in[base[i] + 3];
    P.vv[i] = (const float*)d_in[base[i] + 4];
  }
  P.score_W = (const float*)d_in[16];
  P.score_b = (const float*)d_in[17];
  P.ln_g = (const float*)d_in[18];
  P.ln_b = (const float*)d_in[19];
  P.ch16 = (_Float16*)alloc((size_t)2 * 96 * NVOX * 2);
  P.sp16 = (_Float16*)alloc((size_t)2 * 96 * NVOX * 2);
  P.wqP = (_Float16*)alloc(9216 * 2);
  P.wvP = (_Float16*)alloc(9216 * 2);
  P.sqP = (_Float16*)alloc(9216 * 2);
  P.skP = (_Float16*)alloc(9216 * 2);
  P.svP = (_Float16*)alloc(9216 * 2);
  P.w3P = (_Float16*)alloc(248832 * 2);
  P.sb  = (float*)alloc(7 * 192 * 4);
  P.red = (float*)alloc(514 * 4);
  P.out = (float*)d_out;

  k0_prep<<<1, 256, 0, stream>>>(P);
  k1_chq<<<2 * NTILE, 256, 0, stream>>>(P);
  k2_gate<<<2, 128, 0, stream>>>(P);
  k3_chv<<<2 * NTILE, 256, 0, stream>>>(P);
  k5_smax<<<2, 32, 0, stream>>>(P);
  k6_sp<<<2 * NTILE, 256, 0, stream>>>(P);
  k7_conv3<<<2 * 48 * 24, 192, 0, stream>>>(P);
}
